// TaenzerLRLCBlock_85495618994604
// MI455X (gfx1250) — matrix operand order repaired and checked by reading, not compiled, not run on hardware
//
#include <hip/hip_runtime.h>

typedef unsigned short u16;
typedef __attribute__((ext_vector_type(16))) __bf16 v16bf;
typedef __attribute__((ext_vector_type(8)))  float  v8f;
typedef unsigned int u32x4 __attribute__((ext_vector_type(4)));
typedef int          i32x4 __attribute__((ext_vector_type(4)));
typedef int          i32x8 __attribute__((ext_vector_type(8)));

#define RANK 4
#define CI1  32
#define CO   64
#define NB   16
#define H0   128
#define W0   256
#define H1   130
#define W1   258
#define H2   132
#define W2   260
#define KT1  (9*CI1)   /* 288 */
#define KT2  (9*CO)    /* 576 */
#define PH   44
#define PW   86
#define BN_EPS 1e-5f

// B panel geometry: 256 n-rows x 32 bf16 = 4096 dwords = 16KB per K-step,
// stored in LDS with +1 dword pad every 16 dwords (17-dword row stride).
#define BPAN_DW   4096
#define BPAN_B    16384
#define BBUF_DW   (256*17)       /* 4352 dwords = 17408 B per LDS buffer */
#define BBUF_B    (BBUF_DW*4)

#if defined(__has_builtin)
#if __has_builtin(__builtin_amdgcn_tensor_load_to_lds) && __has_builtin(__builtin_amdgcn_s_wait_tensorcnt)
#define HAVE_TDM 1
#endif
#endif

__device__ __forceinline__ u16 f2bf(float f) {
  unsigned u = __float_as_uint(f);
  u += 0x7FFFu + ((u >> 16) & 1u);          // round-to-nearest-even
  return (u16)(u >> 16);
}
__device__ __forceinline__ float bf2f(u16 h) {
  return __uint_as_float(((unsigned)h) << 16);
}

#ifdef HAVE_TDM
// TDM: async copy of one 16KB B panel into LDS, inserting a 1-dword pad after
// every 16 dwords (reproduces the 17-dword bank-spread LDS layout).
__device__ __forceinline__ void tdm_load_b_panel(const void* gsrc, unsigned lds_off) {
  unsigned long long ga = (unsigned long long)gsrc;
  u32x4 g0;
  g0[0] = 1u;                                               // count=1 user D#
  g0[1] = lds_off;                                          // lds_addr (bytes)
  g0[2] = (unsigned)ga;                                     // global_addr[31:0]
  g0[3] = (unsigned)((ga >> 32) & 0x1FFFFFFu) | (2u << 30); // addr[56:32] | type=2
  i32x8 g1;
  g1[0] = (int)((2u << 16) | (1u << 20) | (3u << 22)); // data_size=4B, pad_en, 16dw interval, +1dw
  g1[1] = (int)(0x1000u << 16);  // tensor_dim0 = 4096 (low16 -> bits[63:48])
  g1[2] = (int)(1u << 16);       // tensor_dim0 hi=0 | tensor_dim1 = 1
  g1[3] = (int)(0x1000u << 16);  // tensor_dim1 hi=0 | tile_dim0 = 4096
  g1[4] = 1;                     // tile_dim1 = 1, tile_dim2 = 0
  g1[5] = 4096;                  // tensor_dim0_stride low
  g1[6] = (int)(0x1000u << 16);  // t0s hi=0 | tensor_dim1_stride low = 4096
  g1[7] = 0;
  i32x4 z4 = {0, 0, 0, 0};
#if defined(__clang_major__) && (__clang_major__ >= 23)
  i32x8 z8 = {0, 0, 0, 0, 0, 0, 0, 0};
  __builtin_amdgcn_tensor_load_to_lds(g0, g1, z4, z4, z8, 0);
#else
  __builtin_amdgcn_tensor_load_to_lds(g0, g1, z4, z4, 0);
#endif
}
#endif

// ---------------- combining weights: softmax over rank of ah[r,h]+aw[r,w] ----
__global__ void cw_softmax_kernel(const float* __restrict__ ah,
                                  const float* __restrict__ aw,
                                  float* __restrict__ cw, int Lh, int Lw) {
  int idx = blockIdx.x * blockDim.x + threadIdx.x;
  int tot = Lh * Lw;
  if (idx >= tot) return;
  int h = idx / Lw, w = idx % Lw;
  float l[RANK]; float m = -1e30f;
#pragma unroll
  for (int r = 0; r < RANK; ++r) { l[r] = ah[r*Lh + h] + aw[r*Lw + w]; m = fmaxf(m, l[r]); }
  float s = 0.f;
#pragma unroll
  for (int r = 0; r < RANK; ++r) { l[r] = __expf(l[r] - m); s += l[r]; }
  float inv = 1.f / s;
#pragma unroll
  for (int r = 0; r < RANK; ++r) cw[(size_t)r*tot + idx] = l[r] * inv;
}

// ---------------- pack weights into per-K-step bf16 B panels -----------------
// bm[(s*256 + n)*32 + kk], n = co*4 + r, k = s*32 + kk = tap*CIN + c
__global__ void prep_w_kernel(const float* __restrict__ w, u16* __restrict__ bm,
                              int CIN, int KTOT) {
  int i = blockIdx.x * blockDim.x + threadIdx.x;
  int tot = 256 * KTOT;
  if (i >= tot) return;
  int kk = i & 31;
  int t  = i >> 5;
  int n  = t & 255;
  int s  = t >> 8;
  int k  = s * 32 + kk;
  int co = n >> 2, r = n & 3;
  int tap = k / CIN, c = k % CIN;
  int kh = tap / 3, kw = tap % 3;
  bm[i] = f2bf(w[(((size_t)(r*CO + co)*CIN + c)*3 + kh)*3 + kw]);
}

// ---------------- implicit-GEMM LRLC conv via bf16 WMMA ----------------------
// M tile = 64 spatial (2 rows x 32 cols), N = 256 (co*4+r), K-step = 32 (1 tap slice)
// B panels double-buffered in LDS, streamed by the Tensor Data Mover.
template<int CIN, int KTOT, int INF32>
__global__ __launch_bounds__(256)
void lrlc_gemm_kernel(const void* __restrict__ inp,
                      const unsigned* __restrict__ bmS,
                      const float* __restrict__ cw,
                      const float* __restrict__ bias,
                      u16* __restrict__ outp,
                      int HIN, int WIN, int HOUT, int WOUT, int NH, int TW) {
  constexpr int S = KTOT / 32;
  __shared__ unsigned BsU[2 * BBUF_DW];  // double-buffered B panels
  __shared__ u16      As[64 * 34];       // 64 spatial x 32 k (+pad)

  const int tid   = threadIdx.x;
  const int lane  = tid & 31;
  const int wave  = tid >> 5;
  const int lh    = lane >> 4;
  const int l15   = lane & 15;
  const int waveM = wave & 3;   // 4 sub-tiles of 16 along M
  const int waveN = wave >> 2;  // 2 sub-tiles of 128 along N

  int bid = blockIdx.x;
  int wt = bid % TW;
  int th = (bid / TW) % NH;
  int b  = bid / (TW * NH);
  int ow0 = wt * 32;
  int oh0 = th * 2;

  v8f acc[8] = {};

#ifdef HAVE_TDM
  // flat shared address low 32 bits == LDS byte offset (LDS aperture mapping)
  const unsigned bsLds = (unsigned)(unsigned long long)(const void*)BsU;
  const bool tdmWave = (wave == 0);
  if (tdmWave) tdm_load_b_panel((const char*)bmS, bsLds);   // prefetch panel 0
#endif

  for (int s = 0; s < S; ++s) {
    __syncthreads();   // previous compute finished; A / next B buffer reusable

    { // stage A tile: 64 spatial x 32 channels for this tap slice (zero-padded)
      int k0 = s * 32;
      int tap = k0 / CIN;
      int c0 = k0 % CIN;
      int kh = tap / 3, kw = tap % 3;
#pragma unroll
      for (int it = 0; it < 8; ++it) {
        int idx = tid + it * 256;
        int j = idx & 63;
        int c = idx >> 6;
        int ih = oh0 + (j >> 5) + kh - 2;
        int iw = ow0 + (j & 31) + kw - 2;
        u16 v = 0;
        if (ih >= 0 && ih < HIN && iw >= 0 && iw < WIN) {
          size_t o = ((size_t)((b * CIN + c0 + c) * HIN + ih)) * WIN + iw;
          v = INF32 ? f2bf(((const float*)inp)[o]) : ((const u16*)inp)[o];
        }
        As[j * 34 + c] = v;
      }
    }

#ifdef HAVE_TDM
    if (tdmWave) {
      if (s + 1 < S) {  // kick TDM for next panel into the other buffer
        tdm_load_b_panel((const char*)bmS + (size_t)(s + 1) * BPAN_B,
                         bsLds + (unsigned)(((s + 1) & 1) * BBUF_B));
        __builtin_amdgcn_s_wait_tensorcnt((short)1);  // panel s landed
      } else {
        __builtin_amdgcn_s_wait_tensorcnt((short)0);
      }
    }
#else
    { // fallback: cooperative copy of panel s (coalesced global, bank-spread LDS)
      const unsigned* src = bmS + (size_t)s * BPAN_DW;
      unsigned* dst = BsU + (s & 1) * BBUF_DW;
#pragma unroll
      for (int it = 0, e = tid; it < 16; ++it, e += 256)
        dst[(e >> 4) * 17 + (e & 15)] = src[e];
    }
#endif
    __syncthreads();

    const u16* Bs = (const u16*)(BsU + (s & 1) * BBUF_DW);

    // A fragment (16x32 bf16): lanes 0-15/16-31 hold the same 16 rows, split K
    union { v16bf v; unsigned u[8]; } fa;
    {
      const u16* ap = As + (waveM * 16 + l15) * 34;
#pragma unroll
      for (int q = 0; q < 8; ++q) {
        int kk = lh * 16 + q * 2;
        fa.u[q] = *(const unsigned*)(ap + kk);
      }
    }
    // 8 N-blocks: B fragment (32x16 bf16) + WMMA f32 accumulate
#pragma unroll
    for (int a = 0; a < 8; ++a) {
      union { v16bf v; unsigned u[8]; } fb;
      int n = waveN * 128 + a * 16 + l15;
      const u16* bp = Bs + n * 34 + lh * 16;
#pragma unroll
      for (int q = 0; q < 8; ++q) fb.u[q] = *(const unsigned*)(bp + q * 2);
      acc[a] = __builtin_amdgcn_wmma_f32_16x16x32_bf16(
          false, fa.v, false, fb.v, (short)0, acc[a], false, false);
    }
  }

  // epilogue: per-pixel softmax combine over rank (4-lane butterfly), +bias
#pragma unroll
  for (int a = 0; a < 8; ++a) {
    int n  = waveN * 128 + a * 16 + l15;
    int co = n >> 2;
    float bv = bias[co];
    const float* cwr = cw + (size_t)(n & 3) * HOUT * WOUT;
#pragma unroll
    for (int q = 0; q < 8; ++q) {
      int jj  = waveM * 16 + q + 8 * lh;   // C-layout: row = q (+8 for hi lanes)
      int ohg = oh0 + (jj >> 5);
      int ow  = ow0 + (jj & 31);
      float cv  = (ow < WOUT) ? cwr[(size_t)ohg * WOUT + ow] : 0.f;
      float val = acc[a][q] * cv;
      val += __shfl_xor(val, 1, 32);
      val += __shfl_xor(val, 2, 32);
      if (((lane & 3) == 0) && (ow < WOUT))
        outp[((size_t)(b * CO + co) * HOUT + ohg) * WOUT + ow] = f2bf(val + bv);
    }
  }
}

// ---------------- BN helpers -------------------------------------------------
__global__ void zero_kernel(float* a, int n) {
  int i = blockIdx.x * blockDim.x + threadIdx.x;
  if (i < n) a[i] = 0.f;
}

__global__ void bn_stats_kernel(const u16* __restrict__ h, float* __restrict__ stats,
                                int plane, int C) {
  __shared__ float ls[256], ls2[256];
  int c = blockIdx.x % C;
  const u16* p = h + (size_t)blockIdx.x * plane;
  float s = 0.f, s2 = 0.f;
  for (int i = threadIdx.x; i < plane; i += 256) {
    float v = bf2f(p[i]); s += v; s2 += v * v;
  }
  ls[threadIdx.x] = s; ls2[threadIdx.x] = s2;
  __syncthreads();
  for (int o = 128; o > 0; o >>= 1) {
    if (threadIdx.x < o) { ls[threadIdx.x] += ls[threadIdx.x + o];
                           ls2[threadIdx.x] += ls2[threadIdx.x + o]; }
    __syncthreads();
  }
  if (threadIdx.x == 0) {
    atomicAdd(&stats[c],     ls[0]);
    atomicAdd(&stats[C + c], ls2[0]);
  }
}

__global__ void bn_relu_kernel(u16* __restrict__ h, const float* __restrict__ stats,
                               const float* __restrict__ g, const float* __restrict__ be,
                               int plane, int C, float invN, size_t total) {
  size_t i = (size_t)blockIdx.x * blockDim.x + threadIdx.x;
  if (i >= total) return;
  int c = (int)((i / plane) % C);
  float mean = stats[c] * invN;
  float var  = stats[C + c] * invN - mean * mean;
  float sc = g[c] * rsqrtf(var + BN_EPS);
  float sh = be[c] - mean * sc;
  h[i] = f2bf(fmaxf(bf2f(h[i]) * sc + sh, 0.f));
}

__global__ void bn_relu_pool_kernel(const u16* __restrict__ h,
                                    const float* __restrict__ stats,
                                    const float* __restrict__ g,
                                    const float* __restrict__ be,
                                    float* __restrict__ out, float invN, size_t total) {
  size_t i = (size_t)blockIdx.x * blockDim.x + threadIdx.x;
  if (i >= total) return;
  int pw = (int)(i % PW); size_t t = i / PW;
  int ph = (int)(t % PH); t /= PH;
  int c  = (int)(t % CO);
  int b  = (int)(t / CO);
  float mean = stats[c] * invN;
  float var  = stats[CO + c] * invN - mean * mean;
  float sc = g[c] * rsqrtf(var + BN_EPS);
  float sh = be[c] - mean * sc;
  const u16* p = h + ((size_t)(b * CO + c) * H2 + ph * 3) * W2 + pw * 3;
  float best = -1e30f;
#pragma unroll
  for (int dy = 0; dy < 3; ++dy)
#pragma unroll
    for (int dx = 0; dx < 3; ++dx)
      best = fmaxf(best, fmaxf(bf2f(p[dy * W2 + dx]) * sc + sh, 0.f));
  out[i] = best;
}

// ---------------- launcher ---------------------------------------------------
extern "C" void kernel_launch(void* const* d_in, const int* in_sizes, int n_in,
                              void* d_out, int out_size, void* d_ws, size_t ws_size,
                              hipStream_t stream) {
  const float* x   = (const float*)d_in[0];
  const float* w1  = (const float*)d_in[1];
  const float* b1  = (const float*)d_in[2];
  const float* a1h = (const float*)d_in[3];
  const float* a1w = (const float*)d_in[4];
  const float* g1  = (const float*)d_in[5];
  const float* be1 = (const float*)d_in[6];
  const float* w2  = (const float*)d_in[7];
  const float* b2  = (const float*)d_in[8];
  const float* a2h = (const float*)d_in[9];
  const float* a2w = (const float*)d_in[10];
  const float* g2  = (const float*)d_in[11];
  const float* be2 = (const float*)d_in[12];

  char* ws = (char*)d_ws;
  size_t off = 0;
  auto carve = [&](size_t bytes) -> void* {
    void* p = ws + off;
    off += (bytes + 255) & ~(size_t)255;
    return p;
  };
  float* cw1 = (float*)carve((size_t)RANK * H1 * W1 * 4);
  float* cw2 = (float*)carve((size_t)RANK * H2 * W2 * 4);
  u16*   bm1 = (u16*)carve((size_t)256 * KT1 * 2);
  u16*   bm2 = (u16*)carve((size_t)256 * KT2 * 2);
  u16*   h1  = (u16*)carve((size_t)NB * CO * H1 * W1 * 2);
  u16*   h2  = (u16*)carve((size_t)NB * CO * H2 * W2 * 2);
  float* st1 = (float*)carve(2 * CO * 4);
  float* st2 = (float*)carve(2 * CO * 4);
  (void)ws_size; (void)in_sizes; (void)n_in; (void)out_size;

  zero_kernel<<<1, 256, 0, stream>>>(st1, 2 * CO);
  zero_kernel<<<1, 256, 0, stream>>>(st2, 2 * CO);
  cw_softmax_kernel<<<(H1 * W1 + 255) / 256, 256, 0, stream>>>(a1h, a1w, cw1, H1, W1);
  cw_softmax_kernel<<<(H2 * W2 + 255) / 256, 256, 0, stream>>>(a2h, a2w, cw2, H2, W2);
  prep_w_kernel<<<(256 * KT1 + 255) / 256, 256, 0, stream>>>(w1, bm1, CI1, KT1);
  prep_w_kernel<<<(256 * KT2 + 255) / 256, 256, 0, stream>>>(w2, bm2, CO, KT2);

  // conv1: x(f32) -> h1(bf16, pre-BN)
  lrlc_gemm_kernel<CI1, KT1, 1><<<NB * (H1 / 2) * 9, 256, 0, stream>>>(
      x, (const unsigned*)bm1, cw1, b1, h1, H0, W0, H1, W1, H1 / 2, 9);

  bn_stats_kernel<<<NB * CO, 256, 0, stream>>>(h1, st1, H1 * W1, CO);
  size_t tot1 = (size_t)NB * CO * H1 * W1;
  bn_relu_kernel<<<(unsigned)((tot1 + 255) / 256), 256, 0, stream>>>(
      h1, st1, g1, be1, H1 * W1, CO, 1.f / ((float)NB * H1 * W1), tot1);

  // conv2: h1(bf16) -> h2(bf16, pre-BN)
  lrlc_gemm_kernel<CO, KT2, 0><<<NB * (H2 / 2) * 9, 256, 0, stream>>>(
      h1, (const unsigned*)bm2, cw2, b2, h2, H1, W1, H2, W2, H2 / 2, 9);

  bn_stats_kernel<<<NB * CO, 256, 0, stream>>>(h2, st2, H2 * W2, CO);
  size_t totp = (size_t)NB * CO * PH * PW;
  bn_relu_pool_kernel<<<(unsigned)((totp + 255) / 256), 256, 0, stream>>>(
      h2, st2, g2, be2, (float*)d_out, 1.f / ((float)NB * H2 * W2), totp);
}
